// TGNetwork_50130858279318
// MI455X (gfx1250) — matrix operand order repaired and checked by reading, not compiled, not run on hardware
//
#include <hip/hip_runtime.h>
#include <hip/hip_bf16.h>

// MI455X / gfx1250, wave32. All GEMMs via v_wmma_f32_16x16x32_bf16.
// B=16384, D=512, G=8, S=64, K=64, E=8, H=256. Expert 0 only reaches output.

typedef __attribute__((ext_vector_type(16))) __bf16 v16bf;
typedef __attribute__((ext_vector_type(8)))  __bf16 v8bf;
typedef __attribute__((ext_vector_type(8)))  float  v8f;

#define BN_EPS 1e-5f

#define WMMA_BF16(a, b, c) \
  __builtin_amdgcn_wmma_f32_16x16x32_bf16(false, (a), false, (b), (short)0, (c), false, false)

// ---------------------------------------------------------------------------
// WMMA fragment loaders (row-major memory).
// A (16x32, 16-bit): lane<16 holds row m=lane, K 0..15 contiguous;
//                    lane>=16 holds row m=lane-16, K 16..31.
// B (32x16, 16-bit): lane<16 holds col n=lane (= weight row n), K 0..15
//                    contiguous; lane>=16 same col, K 16..31.
// ---------------------------------------------------------------------------
__device__ __forceinline__ v16bf load_a_frag(const __bf16* __restrict__ base,
                                             int lda, int k0) {
  const int lane = threadIdx.x & 31;
  const int m    = lane & 15;
  const int klo  = (lane >> 4) ? 16 : 0;
  const __bf16* p = base + (size_t)m * lda + k0 + klo;
  v8bf lo = *(const v8bf*)p;
  v8bf hi = *(const v8bf*)(p + 8);
  return __builtin_shufflevector(lo, hi, 0,1,2,3,4,5,6,7,8,9,10,11,12,13,14,15);
}

__device__ __forceinline__ v16bf load_b_frag(const __bf16* __restrict__ wbase,
                                             int ldb, int k0) {
  const int lane = threadIdx.x & 31;
  const int n    = lane & 15;
  const int koff = (lane >> 4) ? 16 : 0;
  return *(const v16bf*)(wbase + (size_t)n * ldb + k0 + koff);
}

// ---------------------------------------------------------------------------
// Prep kernels: fold eval-mode BN into bf16 weights + fp32 bias.
//   y = relu( (W x - m) * g*rsqrt(v+eps) + b ) = relu( W' x + b' )
// ---------------------------------------------------------------------------
__global__ void fold_bn_kernel(const float* __restrict__ W,
                               const float* __restrict__ g,
                               const float* __restrict__ b,
                               const float* __restrict__ m,
                               const float* __restrict__ v,
                               __bf16* __restrict__ Wb,
                               float* __restrict__ bias,
                               int O, int I) {
  const int stride = gridDim.x * blockDim.x;
  const int tid = blockIdx.x * blockDim.x + threadIdx.x;
  const int total = O * I;
  for (int i = tid; i < total; i += stride) {
    int o = i / I;
    float sc = g[o] * rsqrtf(v[o] + BN_EPS);
    Wb[i] = (__bf16)(W[i] * sc);
  }
  for (int o = tid; o < O; o += stride) {
    float sc = g[o] * rsqrtf(v[o] + BN_EPS);
    bias[o] = b[o] - m[o] * sc;
  }
}

// Build We1'[g][h][s] = We1[0][h][s] * eg[0][g][h]*rsqrt(ev[0][g][h]+eps),
// biasE[g][h] = eb[0][g][h] - em[0][g][h]*scale, and We2b = bf16(We2[0]).
__global__ void expert_prep_kernel(const float* __restrict__ We1,   // [H=256][S=64] (expert 0)
                                   const float* __restrict__ We2,   // [K=64][H=256] (expert 0)
                                   const float* __restrict__ eg,    // [G=8][H=256] (expert 0)
                                   const float* __restrict__ eb,
                                   const float* __restrict__ em,
                                   const float* __restrict__ ev,
                                   __bf16* __restrict__ We1p,       // [8][256][64]
                                   float* __restrict__ biasE,       // [8][256]
                                   __bf16* __restrict__ We2b) {     // [64][256]
  const int stride = gridDim.x * blockDim.x;
  const int tid = blockIdx.x * blockDim.x + threadIdx.x;
  for (int i = tid; i < 8 * 256 * 64; i += stride) {
    int gg = i >> 14;
    int h  = (i >> 6) & 255;
    int s  = i & 63;
    float sc = eg[gg * 256 + h] * rsqrtf(ev[gg * 256 + h] + BN_EPS);
    We1p[i] = (__bf16)(We1[h * 64 + s] * sc);
  }
  for (int i = tid; i < 8 * 256; i += stride) {
    float sc = eg[i] * rsqrtf(ev[i] + BN_EPS);
    biasE[i] = eb[i] - em[i] * sc;
  }
  for (int i = tid; i < 64 * 256; i += stride) {
    We2b[i] = (__bf16)We2[i];
  }
}

__global__ void cvt_bf16_kernel(const float* __restrict__ x,
                                __bf16* __restrict__ y, int n) {
  const int stride = gridDim.x * blockDim.x;
  for (int i = blockIdx.x * blockDim.x + threadIdx.x; i < n; i += stride)
    y[i] = (__bf16)x[i];
}

// ---------------------------------------------------------------------------
// Fused Linear + BN + ReLU GEMM:  Y[B,O] = relu(X[B,I] @ W[O,I]^T + bias[O])
// 256 threads = 8 waves. Block tile: 128 rows x 64 cols; each wave owns a
// 16x64 strip (4 accumulator tiles), K stepped by 32. Weights stay L2-hot.
// ---------------------------------------------------------------------------
__global__ void __launch_bounds__(256)
lbr_gemm_kernel(const __bf16* __restrict__ X, const __bf16* __restrict__ W,
                const float* __restrict__ bias, __bf16* __restrict__ Y,
                int I, int O) {
  const int lane = threadIdx.x & 31;
  const int wave = threadIdx.x >> 5;
  const int row0 = blockIdx.x * 128 + wave * 16;
  const int col0 = blockIdx.y * 64;

  const __bf16* Xb = X + (size_t)row0 * I;

  v8f acc[4] = {};
  for (int k0 = 0; k0 < I; k0 += 32) {
    v16bf a = load_a_frag(Xb, I, k0);
#pragma unroll
    for (int t = 0; t < 4; ++t) {
      v16bf b = load_b_frag(W + (size_t)(col0 + t * 16) * I, I, k0);
      acc[t] = WMMA_BF16(a, b, acc[t]);
    }
  }

  const int n_lane = lane & 15;
  const int m_hi   = (lane >> 4) * 8;   // C/D: vgpr j -> M=j (lanes 0-15), M=j+8 (lanes 16-31)
#pragma unroll
  for (int t = 0; t < 4; ++t) {
    int n = col0 + t * 16 + n_lane;
    float bz = bias[n];
#pragma unroll
    for (int j = 0; j < 8; ++j) {
      int m = m_hi + j;
      float y = acc[t][j] + bz;
      y = y > 0.0f ? y : 0.0f;
      Y[(size_t)(row0 + m) * O + n] = (__bf16)y;
    }
  }
}

// ---------------------------------------------------------------------------
// Fused expert (expert 0): per block = (16 rows, one g):
//   x1[16,256] = enc @ We1'[g]^T  -> bias+ReLU -> LDS (bf16)
//   x2[16,64]  = x1 @ We2^T (K=256 from LDS)
//   out = step( clip(1.2*sigmoid(x2)-0.1, 0, 1) > 0.49999 )  (fp32 0/1)
// 128 threads = 4 waves. Phase 1: wave w does cols [64w,64w+64).
// Phase 2: wave w does output n-tile w (16 cols).
// ---------------------------------------------------------------------------
__global__ void __launch_bounds__(128)
expert_kernel(const __bf16* __restrict__ act4,   // [B][512] = [B][G=8][S=64]
              const __bf16* __restrict__ We1p,   // [8][256][64]
              const float*  __restrict__ biasE,  // [8][256]
              const __bf16* __restrict__ We2b,   // [64][256]
              float* __restrict__ out) {         // [B][8][64]
  __shared__ __bf16 lds[16 * 256];               // 8 KB of the 320 KB WGP LDS

  const int lane = threadIdx.x & 31;
  const int wave = threadIdx.x >> 5;             // 0..3
  const int row0 = blockIdx.x * 16;
  const int g    = blockIdx.y;                   // 0..7

  const __bf16* enc = act4 + (size_t)row0 * 512 + g * 64;  // lda = 512
  const __bf16* W1g = We1p + (size_t)g * 256 * 64;         // [256][64]
  const float*  bE  = biasE + g * 256;

  // -------- phase 1: x1 tile, K = 64 --------
  {
    v8f acc[4] = {};
    const int h0 = wave * 64;
    for (int k0 = 0; k0 < 64; k0 += 32) {
      v16bf a = load_a_frag(enc, 512, k0);
#pragma unroll
      for (int t = 0; t < 4; ++t) {
        v16bf b = load_b_frag(W1g + (size_t)(h0 + t * 16) * 64, 64, k0);
        acc[t] = WMMA_BF16(a, b, acc[t]);
      }
    }
    const int n_lane = lane & 15;
    const int m_hi   = (lane >> 4) * 8;
#pragma unroll
    for (int t = 0; t < 4; ++t) {
      int h = h0 + t * 16 + n_lane;
      float bz = bE[h];
#pragma unroll
      for (int j = 0; j < 8; ++j) {
        int m = m_hi + j;
        float y = acc[t][j] + bz;
        y = y > 0.0f ? y : 0.0f;
        lds[m * 256 + h] = (__bf16)y;
      }
    }
  }
  __syncthreads();

  // -------- phase 2: out tile, K = 256 from LDS --------
  {
    v8f acc = {};
    const int kout0  = wave * 16;
    const int n_lane = lane & 15;
    const int m_a    = lane & 15;
    const int klo    = (lane >> 4) * 16;
    const int koffB  = (lane >> 4) * 16;
    for (int k0 = 0; k0 < 256; k0 += 32) {
      const __bf16* pa = &lds[m_a * 256 + k0 + klo];
      v8bf lo = *(const v8bf*)pa;
      v8bf hi = *(const v8bf*)(pa + 8);
      v16bf a = __builtin_shufflevector(lo, hi,
                  0,1,2,3,4,5,6,7,8,9,10,11,12,13,14,15);
      v16bf b = *(const v16bf*)(We2b + (size_t)(kout0 + n_lane) * 256 + k0 + koffB);
      acc = WMMA_BF16(a, b, acc);
    }
    const int m_hi = (lane >> 4) * 8;
    const int n    = kout0 + n_lane;
#pragma unroll
    for (int j = 0; j < 8; ++j) {
      int m = m_hi + j;
      float x = acc[j];
      float s = 1.0f / (1.0f + __expf(-x));
      float v = 1.2f * s - 0.1f;
      v = fminf(fmaxf(v, 0.0f), 1.0f);
      out[(size_t)(row0 + m) * 512 + g * 64 + n] = (v > 0.49999f) ? 1.0f : 0.0f;
    }
  }
}

// ---------------------------------------------------------------------------
// Host launcher
// ---------------------------------------------------------------------------
extern "C" void kernel_launch(void* const* d_in, const int* in_sizes, int n_in,
                              void* d_out, int out_size, void* d_ws, size_t ws_size,
                              hipStream_t stream) {
  (void)in_sizes; (void)n_in; (void)out_size; (void)ws_size;

  const float* prompt = (const float*)d_in[0];
  const float* W1 = (const float*)d_in[1];
  const float* g1 = (const float*)d_in[2];
  const float* b1 = (const float*)d_in[3];
  const float* m1 = (const float*)d_in[4];
  const float* v1 = (const float*)d_in[5];
  const float* W2 = (const float*)d_in[6];
  const float* g2 = (const float*)d_in[7];
  const float* b2 = (const float*)d_in[8];
  const float* m2 = (const float*)d_in[9];
  const float* v2 = (const float*)d_in[10];
  const float* W3 = (const float*)d_in[11];
  const float* g3 = (const float*)d_in[12];
  const float* b3 = (const float*)d_in[13];
  const float* m3 = (const float*)d_in[14];
  const float* v3 = (const float*)d_in[15];
  const float* W4 = (const float*)d_in[16];
  const float* g4 = (const float*)d_in[17];
  const float* b4 = (const float*)d_in[18];
  const float* m4 = (const float*)d_in[19];
  const float* v4 = (const float*)d_in[20];
  const float* We1 = (const float*)d_in[21];  // [8][256][64], use expert 0
  const float* We2 = (const float*)d_in[22];  // [8][64][256], use expert 0
  const float* eg  = (const float*)d_in[23];  // [8][8][256], use expert 0
  const float* eb  = (const float*)d_in[24];
  const float* em  = (const float*)d_in[25];
  const float* ev  = (const float*)d_in[26];

  const int B = 16384;

  // Workspace carving (256B aligned). ~55 MB with aliasing:
  // regA = prompt_bf16 -> act2 -> act4 (each dead before the next producer).
  char* wsp = (char*)d_ws;
  size_t off = 0;
  auto carve = [&](size_t bytes) -> void* {
    void* p = (void*)(wsp + off);
    off += (bytes + 255) & ~(size_t)255;
    return p;
  };

  __bf16* W1b  = (__bf16*)carve((size_t)64 * 512 * 2);
  float*  b1f  = (float*)carve(64 * 4);
  __bf16* W2b  = (__bf16*)carve((size_t)512 * 64 * 2);
  float*  b2f  = (float*)carve(512 * 4);
  __bf16* W3b  = (__bf16*)carve((size_t)1024 * 512 * 2);
  float*  b3f  = (float*)carve(1024 * 4);
  __bf16* W4b  = (__bf16*)carve((size_t)512 * 1024 * 2);
  float*  b4f  = (float*)carve(512 * 4);
  __bf16* We1p = (__bf16*)carve((size_t)8 * 256 * 64 * 2);
  float*  bEf  = (float*)carve((size_t)8 * 256 * 4);
  __bf16* We2b = (__bf16*)carve((size_t)64 * 256 * 2);
  __bf16* act1 = (__bf16*)carve((size_t)B * 64 * 2);
  __bf16* act3 = (__bf16*)carve((size_t)B * 1024 * 2);
  __bf16* regA = (__bf16*)carve((size_t)B * 512 * 2);  // prompt_bf / act2 / act4

  // --- prep ---
  cvt_bf16_kernel<<<2048, 256, 0, stream>>>(prompt, regA, B * 512);
  fold_bn_kernel<<<128, 256, 0, stream>>>(W1, g1, b1, m1, v1, W1b, b1f, 64, 512);
  fold_bn_kernel<<<128, 256, 0, stream>>>(W2, g2, b2, m2, v2, W2b, b2f, 512, 64);
  fold_bn_kernel<<<512, 256, 0, stream>>>(W3, g3, b3, m3, v3, W3b, b3f, 1024, 512);
  fold_bn_kernel<<<512, 256, 0, stream>>>(W4, g4, b4, m4, v4, W4b, b4f, 512, 1024);
  expert_prep_kernel<<<256, 256, 0, stream>>>(We1, We2, eg, eb, em, ev,
                                              We1p, bEf, We2b);

  // --- LBR chain (bf16 WMMA GEMMs) ---
  lbr_gemm_kernel<<<dim3(B / 128, 1),  256, 0, stream>>>(regA, W1b, b1f, act1, 512, 64);
  lbr_gemm_kernel<<<dim3(B / 128, 8),  256, 0, stream>>>(act1, W2b, b2f, regA, 64, 512);
  lbr_gemm_kernel<<<dim3(B / 128, 16), 256, 0, stream>>>(regA, W3b, b3f, act3, 512, 1024);
  lbr_gemm_kernel<<<dim3(B / 128, 8),  256, 0, stream>>>(act3, W4b, b4f, regA, 1024, 512);

  // --- fused expert + sigmoid/threshold ---
  expert_kernel<<<dim3(B / 16, 8), 128, 0, stream>>>(regA, We1p, bEf, We2b,
                                                     (float*)d_out);
}
